// MultiHeadAttention_45603962749332
// MI455X (gfx1250) — matrix operand order repaired and checked by reading, not compiled, not run on hardware
//
#include <hip/hip_runtime.h>
#include <hip/hip_bf16.h>
#include <stdint.h>

// ---------- sizes (fixed by the reference) ----------
#define S_LEN 2048
#define EDIM  1024
#define HEADS 16
#define HDIM  64
#define MROWS 4096          // B * S
#define BATCH 2

typedef __attribute__((ext_vector_type(16))) __bf16 v16bf;
typedef __attribute__((ext_vector_type(8)))  float  v8f;

struct __align__(16) U4 { unsigned int x, y, z, w; };

union BF16Frag {
    v16bf v;
    U4 u[2];
    unsigned short h[16];
};

__device__ __forceinline__ unsigned short f2bf(float f) {
    unsigned int u = __builtin_bit_cast(unsigned int, f);
    u += 0x7FFFu + ((u >> 16) & 1u);           // round-to-nearest-even
    return (unsigned short)(u >> 16);
}

__device__ __forceinline__ U4 pack8(const unsigned short* h) {
    U4 r;
    r.x = (unsigned)h[0] | ((unsigned)h[1] << 16);
    r.y = (unsigned)h[2] | ((unsigned)h[3] << 16);
    r.z = (unsigned)h[4] | ((unsigned)h[5] << 16);
    r.w = (unsigned)h[6] | ((unsigned)h[7] << 16);
    return r;
}

// =====================================================================
// GEMM: C[m,n] = sum_k A[m,k] * W[n,k] + bias[n]    (M=4096,N=1024,K=1024)
// AMODE: 0 = A fp32, 1 = A bf16
// OMODE: 0 = bf16 [M,N];  1 = bf16 transposed to [B,H,D,S] (for V);  2 = fp32 [M,N]
// block = 256 threads (8 waves), tile 128x128, K-step 32,
// double-buffered LDS (one barrier per K-step, global loads overlap WMMA)
// =====================================================================
template<int AMODE, int OMODE>
__global__ __launch_bounds__(256) void gemm_wmma_kernel(
    const void* __restrict__ Aptr, const float* __restrict__ W,
    const float* __restrict__ bias, void* __restrict__ Outp)
{
    __shared__ unsigned short a_lds[2][128][40];   // 128 rows x 32 k (+8 pad)
    __shared__ unsigned short w_lds[2][128][40];

    const int t    = threadIdx.x;
    const int m0   = blockIdx.y * 128;
    const int n0   = blockIdx.x * 128;
    const int r    = t >> 1;            // staging row 0..127
    const int seg  = (t & 1) * 16;      // staging k-segment

    const int wid  = t >> 5;
    const int lane = t & 31;
    const int lm   = lane & 15;
    const int lh   = lane >> 4;
    const int wrow = (wid >> 1) * 32;   // wave row offset in tile
    const int wcol = (wid & 1) * 64;    // wave col offset in tile

    // staged-tile registers (raw data; conversion deferred to LDS store)
    float  afl[16], wfl[16];
    U4     abr[2];

    auto load_tile = [&](int k0) {
        if (AMODE == 0) {
            const float* ap = (const float*)Aptr + (size_t)(m0 + r) * EDIM + k0 + seg;
            #pragma unroll
            for (int i = 0; i < 16; ++i) afl[i] = ap[i];
        } else {
            const U4* ap4 = (const U4*)((const unsigned short*)Aptr +
                                        (size_t)(m0 + r) * EDIM + k0 + seg);
            abr[0] = ap4[0]; abr[1] = ap4[1];
        }
        const float* wp = W + (size_t)(n0 + r) * EDIM + k0 + seg;
        #pragma unroll
        for (int i = 0; i < 16; ++i) wfl[i] = wp[i];
    };

    auto store_tile = [&](int bufi) {
        U4* da = (U4*)&a_lds[bufi][r][seg];
        if (AMODE == 0) {
            unsigned short hb[16];
            #pragma unroll
            for (int i = 0; i < 16; ++i) hb[i] = f2bf(afl[i]);
            da[0] = pack8(hb); da[1] = pack8(hb + 8);
        } else {
            da[0] = abr[0]; da[1] = abr[1];
        }
        unsigned short hw[16];
        #pragma unroll
        for (int i = 0; i < 16; ++i) hw[i] = f2bf(wfl[i]);
        U4* dw = (U4*)&w_lds[bufi][r][seg];
        dw[0] = pack8(hw); dw[1] = pack8(hw + 8);
    };

    v8f acc[2][4] = {};

    load_tile(0);
    store_tile(0);

    const int NIT = EDIM / 32;
    #pragma unroll 1
    for (int it = 0; it < NIT; ++it) {
        __syncthreads();                       // buf[it&1] ready for all waves
        if (it + 1 < NIT) load_tile((it + 1) * 32);   // overlap with WMMA below

        const int cur = it & 1;
        BF16Frag af[2], bfg[4];
        #pragma unroll
        for (int mi = 0; mi < 2; ++mi) {
            const unsigned short* ar = &a_lds[cur][wrow + mi * 16 + lm][0];
            af[mi].u[0] = *(const U4*)(ar + lh * 8);        // K lh*8..+7
            af[mi].u[1] = *(const U4*)(ar + 16 + lh * 8);   // K 16+lh*8..+7
        }
        #pragma unroll
        for (int ni = 0; ni < 4; ++ni) {
            const unsigned short* wr = &w_lds[cur][wcol + ni * 16 + lm][0];
            bfg[ni].u[0] = *(const U4*)(wr + lh * 8);       // K lh*8..+7
            bfg[ni].u[1] = *(const U4*)(wr + 16 + lh * 8);
        }
        #pragma unroll
        for (int mi = 0; mi < 2; ++mi)
            #pragma unroll
            for (int ni = 0; ni < 4; ++ni)
                acc[mi][ni] = __builtin_amdgcn_wmma_f32_16x16x32_bf16(
                    false, af[mi].v, false, bfg[ni].v, (short)0, acc[mi][ni],
                    false, false);

        if (it + 1 < NIT) store_tile((it + 1) & 1);
    }

    // ---- epilogue: bias + store ----
    #pragma unroll
    for (int mi = 0; mi < 2; ++mi) {
        #pragma unroll
        for (int ni = 0; ni < 4; ++ni) {
            const int gn = n0 + wcol + ni * 16 + lm;
            const float bv = bias[gn];
            const int mb = m0 + wrow + mi * 16 + lh * 8;
            #pragma unroll
            for (int rr = 0; rr < 8; ++rr) {
                const float v = acc[mi][ni][rr] + bv;
                const int gm = mb + rr;
                if (OMODE == 0) {
                    ((unsigned short*)Outp)[(size_t)gm * EDIM + gn] = f2bf(v);
                } else if (OMODE == 1) {
                    const int bb = gm >> 11, ss = gm & (S_LEN - 1);
                    const int hh = gn >> 6,  dd = gn & (HDIM - 1);
                    ((unsigned short*)Outp)
                        [(((size_t)(bb * HEADS + hh) * HDIM + dd) << 11) + ss] = f2bf(v);
                } else {
                    ((float*)Outp)[(size_t)gm * EDIM + gn] = v;
                }
            }
        }
    }
}

// =====================================================================
// Fused attention: per block = one (b,h) and 16 query rows.
//   scores(16x2048) -> LDS, +rel bias, softmax in LDS,
//   write normalized attn once (coalesced), P@V via WMMA -> ctx bf16.
// grid = (128, 32), block = 256 (8 waves)
// =====================================================================
__global__ __launch_bounds__(256) void attn_kernel(
    const unsigned short* __restrict__ Q,   // [B*S, E] bf16
    const unsigned short* __restrict__ Kb,  // [B*S, E] bf16
    const unsigned short* __restrict__ Vt,  // [B,H,D,S] bf16
    const float* __restrict__ rel_table,    // [257, 16]
    float* __restrict__ attn_out,           // [B,H,S,S]
    unsigned short* __restrict__ ctx)       // [B*S, E] bf16
{
    __shared__ float s_lds[16][2052];       // scores for 16 q-rows (+4 pad)
    __shared__ float rel_lds[257 * HEADS];

    const int t  = threadIdx.x;
    const int bh = blockIdx.y;
    const int b  = bh >> 4;
    const int h  = bh & 15;
    const int q0 = blockIdx.x * 16;

    for (int i = t; i < 257 * HEADS; i += 256) rel_lds[i] = rel_table[i];
    __syncthreads();

    const int wid  = t >> 5;
    const int lane = t & 31;
    const int lm   = lane & 15;
    const int lh   = lane >> 4;

    // ---------------- Phase A: scores = Q K^T / 8 + bias ----------------
    {
        BF16Frag aq[2];
        const unsigned short* qrow =
            Q + ((size_t)(b * S_LEN + q0 + lm)) * EDIM + h * HDIM;
        #pragma unroll
        for (int s = 0; s < 2; ++s) {
            const int d0 = s * 32 + lh * 8;
            aq[s].u[0] = *(const U4*)(qrow + d0);
            aq[s].u[1] = *(const U4*)(qrow + d0 + 16);
        }
        const int kt0 = wid * 16;            // 16 k-tiles per wave
        #pragma unroll 1
        for (int kt = kt0; kt < kt0 + 16; ++kt) {
            // branchless prefetch of next k-tile (no EXEC manipulation)
            const int ktp = (kt + 1 < kt0 + 16) ? (kt + 1) : kt;
            __builtin_prefetch(Kb + ((size_t)(b * S_LEN + ktp * 16 + lm)) * EDIM
                                   + h * HDIM, 0, 1);
            v8f acc = {};
            #pragma unroll
            for (int s = 0; s < 2; ++s) {
                BF16Frag bk;
                const unsigned short* kp =
                    Kb + ((size_t)(b * S_LEN + kt * 16 + lm)) * EDIM
                       + h * HDIM + s * 32 + lh * 8;
                bk.u[0] = *(const U4*)kp;
                bk.u[1] = *(const U4*)(kp + 16);
                acc = __builtin_amdgcn_wmma_f32_16x16x32_bf16(
                    false, aq[s].v, false, bk.v, (short)0, acc, false, false);
            }
            const int n  = kt * 16 + lm;     // key position
            const int mb = lh * 8;
            #pragma unroll
            for (int rr = 0; rr < 8; ++rr) {
                const int m = q0 + mb + rr;  // query position
                int rel = n - m;
                rel = rel < -128 ? -128 : (rel > 128 ? 128 : rel);
                s_lds[mb + rr][n] = acc[rr] * 0.125f
                                  + rel_lds[(rel + 128) * HEADS + h];
            }
        }
    }
    __syncthreads();

    // ---------------- Phase B: softmax rows (16 threads per row) ----------------
    {
        const int row  = t >> 4;
        const int seg2 = (t & 15) * 128;
        float* rp = &s_lds[row][seg2];
        float mx = -3.0e38f;
        for (int j = 0; j < 128; j += 4) {
            float4 v = *(float4*)&rp[j];
            mx = fmaxf(mx, fmaxf(fmaxf(v.x, v.y), fmaxf(v.z, v.w)));
        }
        mx = fmaxf(mx, __shfl_xor(mx, 1, 32));
        mx = fmaxf(mx, __shfl_xor(mx, 2, 32));
        mx = fmaxf(mx, __shfl_xor(mx, 4, 32));
        mx = fmaxf(mx, __shfl_xor(mx, 8, 32));
        float sum = 0.0f;
        for (int j = 0; j < 128; j += 4) {
            float4 v = *(float4*)&rp[j];
            v.x = __expf(v.x - mx); v.y = __expf(v.y - mx);
            v.z = __expf(v.z - mx); v.w = __expf(v.w - mx);
            sum += v.x + v.y + v.z + v.w;
            *(float4*)&rp[j] = v;
        }
        sum += __shfl_xor(sum, 1, 32);
        sum += __shfl_xor(sum, 2, 32);
        sum += __shfl_xor(sum, 4, 32);
        sum += __shfl_xor(sum, 8, 32);
        const float inv = 1.0f / sum;
        for (int j = 0; j < 128; j += 4) {
            float4 v = *(float4*)&rp[j];
            v.x *= inv; v.y *= inv; v.z *= inv; v.w *= inv;
            *(float4*)&rp[j] = v;
        }
    }
    __syncthreads();

    // ---------------- Phase B2 / C (read-only on LDS, run concurrently) ----------------
    if (wid >= 4) {
        // waves 4..7: coalesced write of normalized attn (written exactly once)
        float* __restrict__ dst = attn_out + ((size_t)bh * S_LEN + q0) * S_LEN;
        for (int i = t - 128; i < 16 * 512; i += 128) {
            const int row = i >> 9;
            const int c4  = (i & 511) << 2;
            float4 v = *(float4*)&s_lds[row][c4];
            *(float4*)&dst[(size_t)row * S_LEN + c4] = v;
        }
    } else {
        // waves 0..3: out[16, 64] = P @ V  (each wave one 16x16 d-tile)
        const int dt = wid;
        v8f acc = {};
        const unsigned short* vbase =
            Vt + ((size_t)(bh * HDIM + dt * 16 + lm)) * S_LEN;
        const float* arow_p = &s_lds[lm][0];
        #pragma unroll 2
        for (int k0 = 0; k0 < S_LEN; k0 += 32) {
            BF16Frag ap, bv;
            const float* c0p = arow_p + k0 + lh * 8;
            unsigned short hb[16];
            #pragma unroll
            for (int i = 0; i < 8; ++i) hb[i]     = f2bf(c0p[i]);
            #pragma unroll
            for (int i = 0; i < 8; ++i) hb[8 + i] = f2bf(c0p[16 + i]);
            ap.u[0] = pack8(hb); ap.u[1] = pack8(hb + 8);
            bv.u[0] = *(const U4*)(vbase + k0 + lh * 8);
            bv.u[1] = *(const U4*)(vbase + k0 + 16 + lh * 8);
            acc = __builtin_amdgcn_wmma_f32_16x16x32_bf16(
                false, ap.v, false, bv.v, (short)0, acc, false, false);
        }
        const int gf = h * HDIM + dt * 16 + lm;
        const size_t mrow0 = (size_t)b * S_LEN + q0 + lh * 8;
        #pragma unroll
        for (int rr = 0; rr < 8; ++rr)
            ctx[(mrow0 + rr) * EDIM + gf] = f2bf(acc[rr]);
    }
}

// =====================================================================
extern "C" void kernel_launch(void* const* d_in, const int* in_sizes, int n_in,
                              void* d_out, int out_size, void* d_ws, size_t ws_size,
                              hipStream_t stream) {
    const float* query = (const float*)d_in[0];
    const float* key   = (const float*)d_in[1];
    const float* value = (const float*)d_in[2];
    const float* Wq    = (const float*)d_in[3];
    const float* bq    = (const float*)d_in[4];
    const float* Wk    = (const float*)d_in[5];
    const float* bk    = (const float*)d_in[6];
    const float* Wv    = (const float*)d_in[7];
    const float* bv    = (const float*)d_in[8];
    const float* Wo    = (const float*)d_in[9];
    const float* bo    = (const float*)d_in[10];
    const float* rel   = (const float*)d_in[11];

    float* out  = (float*)d_out;                       // [B,S,E]
    float* attn = out + (size_t)MROWS * EDIM;          // [B,H,S,S]

    unsigned short* Qbf = (unsigned short*)d_ws;       // [B*S, E] bf16
    unsigned short* Kbf = Qbf + (size_t)MROWS * EDIM;
    unsigned short* Vt  = Kbf + (size_t)MROWS * EDIM;  // [B,H,D,S] bf16
    unsigned short* ctx = Vt  + (size_t)MROWS * EDIM;  // [B*S, E] bf16

    dim3 gg(EDIM / 128, MROWS / 128);                  // (8, 32)
    gemm_wmma_kernel<0, 0><<<gg, 256, 0, stream>>>(query, Wq, bq, Qbf);
    gemm_wmma_kernel<0, 0><<<gg, 256, 0, stream>>>(key,   Wk, bk, Kbf);
    gemm_wmma_kernel<0, 1><<<gg, 256, 0, stream>>>(value, Wv, bv, Vt);

    attn_kernel<<<dim3(S_LEN / 16, BATCH * HEADS), 256, 0, stream>>>(
        Qbf, Kbf, Vt, rel, attn, ctx);

    gemm_wmma_kernel<1, 2><<<gg, 256, 0, stream>>>(ctx, Wo, bo, out);
}
